// GlobalRecurrentFeatureModule_55800215110060
// MI455X (gfx1250) — matrix operand order repaired and checked by reading, not compiled, not run on hardware
//
#include <hip/hip_runtime.h>
#include <hip/hip_bf16.h>
#include <stdint.h>

// ---------------- model dimensions ----------------
#define BATCH   8192
#define TSTEPS  128
#define FDIM    64        // real features per step
#define CIN     50
#define HDIM    256
#define COUT    50
#define G3      768       // 3*H (gate width)
#define INGRU   320       // F + H

// ---------------- GRU kernel tiling ----------------
#define BT      64        // batch rows per workgroup
#define NWAVES  8
#define WGSIZE  256

typedef __attribute__((ext_vector_type(16))) __bf16 bf16x16;
typedef __attribute__((ext_vector_type(8)))  float  f32x8;

union ABFrag { bf16x16 v; uint4 q[2]; };

__device__ __forceinline__ unsigned short f2bf(float f) {
  unsigned u = __float_as_uint(f);
  u += 0x7fffu + ((u >> 16) & 1u);            // round-to-nearest-even
  return (unsigned short)(u >> 16);
}
__device__ __forceinline__ float bf2f(unsigned short h) {
  return __uint_as_float(((unsigned)h) << 16);
}
__device__ __forceinline__ float sigmoidf_(float x) {
  return 1.f / (1.f + __expf(-x));
}
__device__ __forceinline__ float tanh_fast(float x) {
  float ax = fabsf(x);
  float e  = __expf(-2.f * ax);
  float t  = (1.f - e) / (1.f + e);
  return copysignf(t, x);
}

__device__ __forceinline__ f32x8 wmma_bf16(const ABFrag& a, const ABFrag& b, f32x8 c) {
  // D = A(16x32 bf16) x B(32x16 bf16) + C(16x16 f32)
  return __builtin_amdgcn_wmma_f32_16x16x32_bf16(false, a.v, false, b.v,
                                                 (short)0, c, false, false);
}

// 2x2 register-blocked K-loop:
//   c{m}{n} += A[m0+16m : +16, 0:K] x Wt[0:K, g0+16n : +16]
// A: bf16 LDS row-major (stride strideA elems).  Wt: bf16 global K-major (row length G3).
// One 64B B run + two 32B A runs feed 4 WMMAs per K-step.
__device__ __forceinline__ void kloop4(f32x8& c00, f32x8& c01, f32x8& c10, f32x8& c11,
                                       const unsigned short* __restrict__ Abase,
                                       int strideA, int K,
                                       const unsigned short* __restrict__ Wt, int g0,
                                       int lane, int m0) {
  ABFrag a0, a1, b0, b1;
  const unsigned short* ap0 = Abase + (m0 + (lane & 15)) * strideA;
  const unsigned short* ap1 = ap0 + 16 * strideA;
  const int asub = (lane >> 4) << 3;          // 0 or 8
  for (int k0 = 0; k0 < K; k0 += 32) {
    const unsigned short* bp = Wt + (size_t)(k0 + asub) * G3 + g0 + (lane & 15);
    #pragma unroll
    for (int s = 0; s < 16; ++s) {
      size_t ko = (size_t)((s & 7) + ((s >> 3) << 4)) * G3;
      b0.v[s] = __builtin_bit_cast(__bf16, bp[ko]);
      b1.v[s] = __builtin_bit_cast(__bf16, bp[ko + 16]);
    }
    a0.q[0] = *(const uint4*)(ap0 + k0 + asub);
    a0.q[1] = *(const uint4*)(ap0 + k0 + 16 + asub);
    a1.q[0] = *(const uint4*)(ap1 + k0 + asub);
    a1.q[1] = *(const uint4*)(ap1 + k0 + 16 + asub);
    c00 = wmma_bf16(a0, b0, c00);
    c01 = wmma_bf16(a0, b1, c01);
    c10 = wmma_bf16(a1, b0, c10);
    c11 = wmma_bf16(a1, b1, c11);
  }
}

// One GRU cell for the workgroup's BT x HDIM tile.
// Phase A: r,z gates (cols 0..511). Phase B: n gate + h update (cols 512..767).
// Jobs are 32-row x 32-col macro-tiles (2x2 of 16x16 WMMA tiles).
__device__ __forceinline__ void gru_cell(
    const unsigned short* __restrict__ Ain, int strideA, int Kin,
    const unsigned short* __restrict__ WiT,        // [Kin][G3] bf16 K-major
    const float* __restrict__ catproj,             // non-null => gi init from catproj [B,G3]
    const float* __restrict__ b_ih,                // used when catproj == null
    const unsigned short* __restrict__ Hprev,      // [BT][HDIM] bf16 LDS
    const unsigned short* __restrict__ WhT,        // [HDIM][G3] bf16 K-major
    const float* __restrict__ b_hh,
    unsigned short* __restrict__ Hnext,            // [BT][HDIM] bf16 LDS
    float* __restrict__ rbuf, float* __restrict__ zbuf,
    int b0, int lane, int wave)
{
  // ---- phase A: r and z (gate cols 0..511) ----
  for (int j = wave; j < 32; j += NWAVES) {     // 2 m-pairs x 16 n-pairs
    int m0  = (j >> 4) << 5;                    // 0 or 32
    int ntp = j & 15;                           // 32-col macro tile
    int g0  = ntp << 5;
    int gc  = g0 + (lane & 15);
    int rb  = m0 + ((lane >> 4) << 3);          // local output row base (C layout)
    f32x8 ai00, ai01, ai10, ai11, ah00, ah01, ah10, ah11;
    float bh0 = b_hh[gc], bh1 = b_hh[gc + 16];
    if (catproj) {
      const float* cp = catproj + (size_t)(b0 + rb) * G3 + gc;
      #pragma unroll
      for (int v = 0; v < 8; ++v) {
        ai00[v] = cp[(size_t)v * G3];
        ai01[v] = cp[(size_t)v * G3 + 16];
        ai10[v] = cp[(size_t)(v + 16) * G3];
        ai11[v] = cp[(size_t)(v + 16) * G3 + 16];
        ah00[v] = bh0; ah01[v] = bh1; ah10[v] = bh0; ah11[v] = bh1;
      }
    } else {
      float bi0 = b_ih[gc], bi1 = b_ih[gc + 16];
      #pragma unroll
      for (int v = 0; v < 8; ++v) {
        ai00[v] = bi0; ai01[v] = bi1; ai10[v] = bi0; ai11[v] = bi1;
        ah00[v] = bh0; ah01[v] = bh1; ah10[v] = bh0; ah11[v] = bh1;
      }
    }
    kloop4(ai00, ai01, ai10, ai11, Ain,   strideA, Kin,  WiT, g0, lane, m0);
    kloop4(ah00, ah01, ah10, ah11, Hprev, HDIM,    HDIM, WhT, g0, lane, m0);
    float* dst = (ntp < 8) ? rbuf : zbuf;       // r: cols 0..255, z: cols 256..511
    int c0 = ((ntp & 7) << 5) + (lane & 15);
    #pragma unroll
    for (int v = 0; v < 8; ++v) {
      dst[(rb + v)      * HDIM + c0]      = sigmoidf_(ai00[v] + ah00[v]);
      dst[(rb + v)      * HDIM + c0 + 16] = sigmoidf_(ai01[v] + ah01[v]);
      dst[(rb + 16 + v) * HDIM + c0]      = sigmoidf_(ai10[v] + ah10[v]);
      dst[(rb + 16 + v) * HDIM + c0 + 16] = sigmoidf_(ai11[v] + ah11[v]);
    }
  }
  __syncthreads();

  // ---- phase B: n gate + h update (gate cols 512..767) ----
  for (int j = wave; j < 16; j += NWAVES) {     // 2 m-pairs x 8 n-pairs
    int m0  = (j >> 3) << 5;
    int ntp = j & 7;
    int g0  = 512 + (ntp << 5);
    int gc  = g0 + (lane & 15);
    int rb  = m0 + ((lane >> 4) << 3);
    f32x8 ai00, ai01, ai10, ai11, ah00, ah01, ah10, ah11;
    float bh0 = b_hh[gc], bh1 = b_hh[gc + 16];
    if (catproj) {
      const float* cp = catproj + (size_t)(b0 + rb) * G3 + gc;
      #pragma unroll
      for (int v = 0; v < 8; ++v) {
        ai00[v] = cp[(size_t)v * G3];
        ai01[v] = cp[(size_t)v * G3 + 16];
        ai10[v] = cp[(size_t)(v + 16) * G3];
        ai11[v] = cp[(size_t)(v + 16) * G3 + 16];
        ah00[v] = bh0; ah01[v] = bh1; ah10[v] = bh0; ah11[v] = bh1;
      }
    } else {
      float bi0 = b_ih[gc], bi1 = b_ih[gc + 16];
      #pragma unroll
      for (int v = 0; v < 8; ++v) {
        ai00[v] = bi0; ai01[v] = bi1; ai10[v] = bi0; ai11[v] = bi1;
        ah00[v] = bh0; ah01[v] = bh1; ah10[v] = bh0; ah11[v] = bh1;
      }
    }
    kloop4(ai00, ai01, ai10, ai11, Ain,   strideA, Kin,  WiT, g0, lane, m0);
    kloop4(ah00, ah01, ah10, ah11, Hprev, HDIM,    HDIM, WhT, g0, lane, m0);
    int c0 = (ntp << 5) + (lane & 15);          // output h column base
    #pragma unroll
    for (int v = 0; v < 8; ++v) {
      #pragma unroll
      for (int q = 0; q < 4; ++q) {
        int row = rb + v + ((q >> 1) << 4);
        int col = c0 + ((q & 1) << 4);
        float aiv = (q == 0) ? ai00[v] : (q == 1) ? ai01[v] : (q == 2) ? ai10[v] : ai11[v];
        float ahv = (q == 0) ? ah00[v] : (q == 1) ? ah01[v] : (q == 2) ? ah10[v] : ah11[v];
        float r  = rbuf[row * HDIM + col];
        float z  = zbuf[row * HDIM + col];
        float hp = bf2f(Hprev[row * HDIM + col]);
        float nn = tanh_fast(aiv + r * ahv);
        Hnext[row * HDIM + col] = f2bf((1.f - z) * nn + z * hp);
      }
    }
  }
  __syncthreads();
}

// -------------------- persistent fused GRU + MLP tail --------------------
__global__ __launch_bounds__(WGSIZE, 1)
void k_gru(const float* __restrict__ x,              // [B,T,F]
           const float* __restrict__ catproj,        // [B,G3] (includes b_ih0)
           const unsigned short* __restrict__ WxT,   // [F][G3]   bf16
           const unsigned short* __restrict__ Whh0T, // [H][G3]   bf16
           const unsigned short* __restrict__ Wih1T, // [H][G3]   bf16
           const unsigned short* __restrict__ Whh1T, // [H][G3]   bf16
           const float* __restrict__ b_hh0,
           const float* __restrict__ b_ih1,
           const float* __restrict__ b_hh1,
           const float* __restrict__ fc21_W, const float* __restrict__ fc21_b,
           const float* __restrict__ fc22_W, const float* __restrict__ fc22_b,
           const float* __restrict__ fc3_W,  const float* __restrict__ fc3_b,
           float* __restrict__ out_real,
           float* __restrict__ out_e)
{
  extern __shared__ __align__(16) char smem[];
  unsigned short* h0a = (unsigned short*)smem;       // BT*HDIM bf16
  unsigned short* h0b = h0a + BT * HDIM;
  unsigned short* h1a = h0b + BT * HDIM;
  unsigned short* h1b = h1a + BT * HDIM;
  unsigned short* xs  = h1b + BT * HDIM;             // BT*FDIM bf16
  float* rbuf = (float*)(xs + BT * FDIM);            // BT*HDIM f32
  float* zbuf = rbuf + BT * HDIM;                    // BT*HDIM f32

  const int tid  = threadIdx.x;
  const int lane = tid & 31;
  const int wave = tid >> 5;
  const int b0   = blockIdx.x * BT;

  for (int i = tid; i < BT * HDIM; i += WGSIZE) { h0a[i] = 0; h1a[i] = 0; }
  __syncthreads();

  unsigned short* h0c = h0a; unsigned short* h0n = h0b;
  unsigned short* h1c = h1a; unsigned short* h1n = h1b;

  for (int t = 0; t < TSTEPS; ++t) {
    // stage x_t tile: f32 global -> bf16 LDS
    for (int i = tid; i < BT * FDIM; i += WGSIZE) {
      int r = i >> 6, c = i & 63;
      xs[i] = f2bf(x[((size_t)(b0 + r) * TSTEPS + t) * FDIM + c]);
    }
    if (t + 1 < TSTEPS)  // warm next step's tile into cache
      __builtin_prefetch(&x[((size_t)(b0 + (tid & 63)) * TSTEPS + t + 1) * FDIM], 0, 1);
    __syncthreads();

    // layer 0: gi = catproj + x_t*WxT ; gh = h0*Whh0T
    gru_cell(xs, FDIM, FDIM, WxT, catproj, nullptr,
             h0c, Whh0T, b_hh0, h0n, rbuf, zbuf, b0, lane, wave);
    // layer 1: gi = b_ih1 + h0_new*Wih1T ; gh = h1*Whh1T
    gru_cell(h0n, HDIM, HDIM, Wih1T, nullptr, b_ih1,
             h1c, Whh1T, b_hh1, h1n, rbuf, zbuf, b0, lane, wave);

    unsigned short* tmp;
    tmp = h0c; h0c = h0n; h0n = tmp;
    tmp = h1c; h1c = h1n; h1n = tmp;
  }

  // ---------------- MLP tail (one-time, scalar f32) ----------------
  // e1 = relu(h1 @ fc21^T + b)  -> rbuf
  for (int i = tid; i < BT * HDIM; i += WGSIZE) {
    int r = i >> 8, c = i & 255;
    float s = fc21_b[c];
    const unsigned short* hp = h1c + r * HDIM;
    const float* wp = fc21_W + (size_t)c * HDIM;
    for (int k = 0; k < HDIM; ++k) s += bf2f(hp[k]) * wp[k];
    rbuf[i] = fmaxf(s, 0.f);
  }
  __syncthreads();
  // e = relu(e1 @ fc22^T + b) -> zbuf + global
  for (int i = tid; i < BT * HDIM; i += WGSIZE) {
    int r = i >> 8, c = i & 255;
    float s = fc22_b[c];
    const float* ep = rbuf + r * HDIM;
    const float* wp = fc22_W + (size_t)c * HDIM;
    for (int k = 0; k < HDIM; ++k) s += ep[k] * wp[k];
    float e = fmaxf(s, 0.f);
    zbuf[i] = e;
    out_e[(size_t)(b0 + r) * HDIM + c] = e;
  }
  __syncthreads();
  // real_out = e @ fc3^T + b
  if (tid < BT) {
    float s = fc3_b[0];
    const float* ep = zbuf + tid * HDIM;
    for (int k = 0; k < HDIM; ++k) s += ep[k] * fc3_W[k];
    out_real[b0 + tid] = s;
  }
}

// -------------------- AE head + prep kernels --------------------
__global__ void k_cat_emb(const float* __restrict__ cat_in,
                          const float* __restrict__ W, const float* __restrict__ b,
                          float* __restrict__ cat_emb) {
  int idx = blockIdx.x * blockDim.x + threadIdx.x;
  if (idx >= BATCH * HDIM) return;
  int bb = idx >> 8, h = idx & 255;
  float s = b[h];
  const float* xi = cat_in + (size_t)bb * CIN;
  const float* wr = W + (size_t)h * CIN;
  #pragma unroll
  for (int c = 0; c < CIN; ++c) s += xi[c] * wr[c];
  cat_emb[idx] = sigmoidf_(s);
}

__global__ void k_cat_recon(const float* __restrict__ cat_emb,
                            const float* __restrict__ W, const float* __restrict__ b,
                            float* __restrict__ out) {
  int lane = threadIdx.x & 31;
  int wave = threadIdx.x >> 5;
  int row  = blockIdx.x * 8 + wave;
  if (row >= BATCH) return;
  const float* e = cat_emb + (size_t)row * HDIM;
  float v0 = -3.4e38f, v1 = -3.4e38f;
  int c0 = lane, c1 = lane + 32;
  if (c0 < COUT) { float s = b[c0]; const float* w = W + (size_t)c0 * HDIM;
                   for (int k = 0; k < HDIM; ++k) s += e[k] * w[k]; v0 = s; }
  if (c1 < COUT) { float s = b[c1]; const float* w = W + (size_t)c1 * HDIM;
                   for (int k = 0; k < HDIM; ++k) s += e[k] * w[k]; v1 = s; }
  float m = fmaxf(v0, v1);
  for (int o = 16; o; o >>= 1) m = fmaxf(m, __shfl_xor(m, o, 32));
  float e0 = (c0 < COUT) ? __expf(v0 - m) : 0.f;
  float e1 = (c1 < COUT) ? __expf(v1 - m) : 0.f;
  float s = e0 + e1;
  for (int o = 16; o; o >>= 1) s += __shfl_xor(s, o, 32);
  float inv = 1.f / s;
  if (c0 < COUT) out[(size_t)row * COUT + c0] = e0 * inv;
  if (c1 < COUT) out[(size_t)row * COUT + c1] = e1 * inv;
}

// Convert recurrent weights f32 row-major -> bf16 K-major (transposed) in ws.
__global__ void k_prep(const float* __restrict__ Wih0, const float* __restrict__ Whh0,
                       const float* __restrict__ Wih1, const float* __restrict__ Whh1,
                       unsigned short* __restrict__ WxT,   unsigned short* __restrict__ Whh0T,
                       unsigned short* __restrict__ Wih1T, unsigned short* __restrict__ Whh1T) {
  int i = blockIdx.x * blockDim.x + threadIdx.x;
  if (i < FDIM * G3) { int k = i / G3, n = i % G3;
    WxT[i] = f2bf(Wih0[(size_t)n * INGRU + HDIM + k]); return; }
  i -= FDIM * G3;
  if (i < HDIM * G3) { int k = i / G3, n = i % G3;
    Whh0T[i] = f2bf(Whh0[(size_t)n * HDIM + k]); return; }
  i -= HDIM * G3;
  if (i < HDIM * G3) { int k = i / G3, n = i % G3;
    Wih1T[i] = f2bf(Wih1[(size_t)n * HDIM + k]); return; }
  i -= HDIM * G3;
  if (i < HDIM * G3) { int k = i / G3, n = i % G3;
    Whh1T[i] = f2bf(Whh1[(size_t)n * HDIM + k]); return; }
}

// catproj[b][n] = b_ih0[n] + cat_emb[b] . W_ih0[n, 0:H]   (f32, one-time)
__global__ void k_catproj(const float* __restrict__ cat_emb,
                          const float* __restrict__ Wih0, const float* __restrict__ b_ih0,
                          float* __restrict__ catproj) {
  size_t idx = (size_t)blockIdx.x * blockDim.x + threadIdx.x;
  if (idx >= (size_t)BATCH * G3) return;
  int bb = (int)(idx / G3), n = (int)(idx % G3);
  float s = b_ih0[n];
  const float* e = cat_emb + (size_t)bb * HDIM;
  const float* w = Wih0 + (size_t)n * INGRU;
  for (int k = 0; k < HDIM; ++k) s += e[k] * w[k];
  catproj[idx] = s;
}

// -------------------- launch --------------------
extern "C" void kernel_launch(void* const* d_in, const int* in_sizes, int n_in,
                              void* d_out, int out_size, void* d_ws, size_t ws_size,
                              hipStream_t stream) {
  const float* real_input = (const float*)d_in[0];
  const float* cat_input  = (const float*)d_in[1];
  const float* ae_in_W  = (const float*)d_in[2];
  const float* ae_in_b  = (const float*)d_in[3];
  const float* ae_out_W = (const float*)d_in[4];
  const float* ae_out_b = (const float*)d_in[5];
  const float* W_ih0 = (const float*)d_in[6];
  const float* W_hh0 = (const float*)d_in[7];
  const float* b_ih0 = (const float*)d_in[8];
  const float* b_hh0 = (const float*)d_in[9];
  const float* W_ih1 = (const float*)d_in[10];
  const float* W_hh1 = (const float*)d_in[11];
  const float* b_ih1 = (const float*)d_in[12];
  const float* b_hh1 = (const float*)d_in[13];
  const float* fc21_W = (const float*)d_in[14];
  const float* fc21_b = (const float*)d_in[15];
  const float* fc22_W = (const float*)d_in[16];
  const float* fc22_b = (const float*)d_in[17];
  const float* fc3_W  = (const float*)d_in[18];
  const float* fc3_b  = (const float*)d_in[19];

  float* out      = (float*)d_out;
  float* out_real = out;                                   // [B,1]
  float* out_e    = out + BATCH;                           // [B,H]
  float* out_rec  = out_e + (size_t)BATCH * HDIM;          // [B,COUT]
  float* out_emb  = out_rec + (size_t)BATCH * COUT;        // [B,H]

  // workspace layout (needs ~26.5 MB)
  char* ws = (char*)d_ws;
  float* catproj = (float*)ws;
  size_t off = (size_t)BATCH * G3 * sizeof(float);
  unsigned short* WxT   = (unsigned short*)(ws + off); off += (size_t)FDIM * G3 * 2;
  unsigned short* Whh0T = (unsigned short*)(ws + off); off += (size_t)HDIM * G3 * 2;
  unsigned short* Wih1T = (unsigned short*)(ws + off); off += (size_t)HDIM * G3 * 2;
  unsigned short* Whh1T = (unsigned short*)(ws + off); off += (size_t)HDIM * G3 * 2;

  hipLaunchKernelGGL(k_cat_emb, dim3((BATCH * HDIM + 255) / 256), dim3(256), 0, stream,
                     cat_input, ae_in_W, ae_in_b, out_emb);

  int prepN = FDIM * G3 + 3 * HDIM * G3;
  hipLaunchKernelGGL(k_prep, dim3((prepN + 255) / 256), dim3(256), 0, stream,
                     W_ih0, W_hh0, W_ih1, W_hh1, WxT, Whh0T, Wih1T, Whh1T);

  hipLaunchKernelGGL(k_cat_recon, dim3(BATCH / 8), dim3(256), 0, stream,
                     out_emb, ae_out_W, ae_out_b, out_rec);

  hipLaunchKernelGGL(k_catproj, dim3((int)(((size_t)BATCH * G3 + 255) / 256)), dim3(256), 0, stream,
                     out_emb, W_ih0, b_ih0, catproj);

  size_t smem = (size_t)(4 * BT * HDIM + BT * FDIM) * 2   // bf16 h0 x2, h1 x2, x
              + (size_t)2 * BT * HDIM * 4;                // f32 r, z   => 270336 B
  hipLaunchKernelGGL(k_gru, dim3(BATCH / BT), dim3(WGSIZE), smem, stream,
                     real_input, catproj, WxT, Whh0T, Wih1T, Whh1T,
                     b_hh0, b_ih1, b_hh1,
                     fc21_W, fc21_b, fc22_W, fc22_b, fc3_W, fc3_b,
                     out_real, out_e);
  (void)in_sizes; (void)n_in; (void)out_size; (void)ws_size;
}
